// FullAttention_8400956031104
// MI455X (gfx1250) — matrix operand order repaired and checked by reading, not compiled, not run on hardware
//
#include <hip/hip_runtime.h>

// CDNA5 / gfx1250 flash-attention:
//   Q,K,V: [B=2, L/S=2048, H=16, E/D=64] fp32 ; out: [B,L,H,D] fp32
//   scores = Q.K^T ; lower-triangle (s<=l) filled with -1e9 ; scale AFTER mask ;
//   softmax ; out = A.V
//
// Optimized path:
//   pre-pass: K -> f16 [b][h][s][e], V -> f16 TRANSPOSED [b][h][d][s] in d_ws
//   main:     block = 8 waves = 128 query rows of one (b,h); shared 32-key
//             K/V^T tiles double-buffered in LDS via global_load_async_to_lds_b128
//             (ASYNCcnt); per-wave WMMA f16 (f32 accum) with online softmax.
//   Triangular skip: key blocks entirely in the masked prefix contribute
//   exactly 0 (alpha underflows to 0.0f once a real score arrives), so each
//   block starts at its own diagonal. The last l-block runs the full range to
//   reproduce the all-masked row (l = L-1 -> uniform softmax over all keys).

typedef __attribute__((ext_vector_type(16))) _Float16 v16h;
typedef __attribute__((ext_vector_type(8)))  float    v8f;

#define LDS_STRIDE 36   // P-transpose row stride in halves (32 + 4 pad)

__device__ __forceinline__ float rowmax16(float v) {
  v = fmaxf(v, __shfl_xor(v, 1, 32));
  v = fmaxf(v, __shfl_xor(v, 2, 32));
  v = fmaxf(v, __shfl_xor(v, 4, 32));
  v = fmaxf(v, __shfl_xor(v, 8, 32));
  return v;
}
__device__ __forceinline__ float rowsum16(float v) {
  v += __shfl_xor(v, 1, 32);
  v += __shfl_xor(v, 2, 32);
  v += __shfl_xor(v, 4, 32);
  v += __shfl_xor(v, 8, 32);
  return v;
}

// ---------------------------------------------------------------------------
// Pre-pass: fp32 -> f16 staging.  Kh[b][h][s][e] ; Vt[b][h][d][s] (transposed)
// ---------------------------------------------------------------------------
__global__ __launch_bounds__(256) void stage_kv_f16(
    const float* __restrict__ K, const float* __restrict__ V,
    _Float16* __restrict__ Kh, _Float16* __restrict__ Vt)
{
  constexpr int S = 2048, H = 16, E = 64;
  const size_t t = (size_t)blockIdx.x * blockDim.x + threadIdx.x; // 0 .. 4Mi-1

  { // Kh: dst [b][h][s][e] <- src [b][s][h][e]
    const int e = (int)(t & 63);
    size_t r = t >> 6;
    const int s = (int)(r & (S - 1)); r >>= 11;
    const int h = (int)(r & (H - 1));
    const int b = (int)(r >> 4);
    Kh[t] = (_Float16)K[(((size_t)b * S + s) * H + h) * E + e];
  }
  { // Vt: dst [b][h][d][s] <- src [b][s][h][d]
    const int s = (int)(t & (S - 1));
    size_t r = t >> 11;
    const int d = (int)(r & 63); r >>= 6;
    const int h = (int)(r & (H - 1));
    const int b = (int)(r >> 4);
    Vt[t] = (_Float16)V[(((size_t)b * S + s) * H + h) * E + d];
  }
}

// ---------------------------------------------------------------------------
// Optimized main kernel: staged f16 K / V^T, async-to-LDS double buffering
// ---------------------------------------------------------------------------
__global__ __launch_bounds__(256) void flash_attn_staged(
    const float* __restrict__ Q, const _Float16* __restrict__ Kh,
    const _Float16* __restrict__ Vt, float* __restrict__ out)
{
  constexpr int L = 2048, H = 16, E = 64, D = 64, S = 2048;
  constexpr float SCALE = 0.125f;
  constexpr float NEGF  = -1.0e9f;

  __shared__ alignas(16) _Float16 sK[2][32 * 64];   // [s_local][e]
  __shared__ alignas(16) _Float16 sV[2][64 * 32];   // [d][s_local]
  __shared__ alignas(16) _Float16 ldsP[8][16 * LDS_STRIDE];

  const int lane = threadIdx.x & 31;
  const int wave = threadIdx.x >> 5;

  const int nLC = L / 128;                       // 16 l-chunks per head
  const int l0b = (blockIdx.x % nLC) * 128;      // block's first query row
  const int l0  = l0b + wave * 16;               // wave's first query row
  const int h   = (blockIdx.x / nLC) % H;
  const int b   =  blockIdx.x / (nLC * H);

  const int laneHi = lane >> 4;
  const int n16    = lane & 15;

  const size_t headIdx = (size_t)b * H + h;
  const _Float16* Kh_head = Kh + headIdx * S * E;         // rows contiguous
  const _Float16* Vt_head = Vt + headIdx * (size_t)D * S; // row d: S halves
  const float*    Qb = Q   + ((size_t)b * L) * H * E + (size_t)h * E;
  float*          Ob = out + ((size_t)b * L) * H * D + (size_t)h * D;

  // ---- Q A-fragments (16x32 f16, two K-halves of E=64) ----
  v16h aQ[2];
  {
    const float* qr = Qb + (size_t)(l0 + n16) * (H * E);
    #pragma unroll
    for (int eh = 0; eh < 2; ++eh) {
      const int e0 = 32 * eh + 16 * laneHi;
      #pragma unroll
      for (int i = 0; i < 8; ++i) aQ[eh][i]     = (_Float16)qr[e0 + i];
      #pragma unroll
      for (int i = 0; i < 8; ++i) aQ[eh][8 + i] = (_Float16)qr[e0 + 8 + i];
    }
  }

  v8f   O[4] = {v8f{}, v8f{}, v8f{}, v8f{}};
  float m_i[8], l_i[8];
  #pragma unroll
  for (int j = 0; j < 8; ++j) { m_i[j] = -3.0e38f; l_i[j] = 0.0f; }

  _Float16* myP = &ldsP[wave][0];
  const int chunk = wave * 32 + lane;   // 0..255 : block-wide copy lane id

  // cooperative async stage of one 32-key block into LDS buffer `buf`
  auto stage = [&](int buf, int s0) {
    { // K tile: 32 rows x 64 halves, contiguous 4 KB; 16 B per lane
      const _Float16* src = Kh_head + (size_t)s0 * 64 + (size_t)chunk * 8;
      unsigned       dst = (unsigned)(size_t)&sK[buf][chunk * 8];
      unsigned long long ga = (unsigned long long)(size_t)src;
      asm volatile("global_load_async_to_lds_b128 %0, %1, off"
                   :: "v"(dst), "v"(ga) : "memory");
    }
    { // V^T tile: 64 rows x 32 halves; 4 x 16 B per row
      const int d = chunk >> 2, part = chunk & 3;
      const _Float16* src = Vt_head + (size_t)d * S + s0 + part * 8;
      unsigned       dst = (unsigned)(size_t)&sV[buf][d * 32 + part * 8];
      unsigned long long ga = (unsigned long long)(size_t)src;
      asm volatile("global_load_async_to_lds_b128 %0, %1, off"
                   :: "v"(dst), "v"(ga) : "memory");
    }
  };

  // Triangular skip: fully-masked prefix blocks contribute exactly zero
  // for every row that later sees a real key. Last l-block runs full range
  // so the all-masked row (l = L-1) reproduces the uniform softmax.
  const int s_start = (l0b == L - 128) ? 0 : l0b;

  stage(0, s_start);
  int cur = 0;

  for (int s0 = s_start; s0 < S; s0 += 32) {
    asm volatile("s_wait_asynccnt 0x0" ::: "memory"); // my copies for `cur` landed
    __syncthreads();                                  // everyone's copies landed
    if (s0 + 32 < S) stage(cur ^ 1, s0 + 32);         // overlap next block's copy

    const _Float16* kT = &sK[cur][0];
    const _Float16* vT = &sV[cur][0];

    // ---- scores: two 16x16 sub-tiles, K=E=64 via two chained WMMAs ----
    v8f sc[2];
    #pragma unroll
    for (int sub = 0; sub < 2; ++sub) {
      const _Float16* kr = kT + (size_t)(sub * 16 + n16) * 64 + 16 * laneHi;
      v16h bK0 = *(const v16h*)kr;          // e = 16*laneHi + 0..15
      v16h bK1 = *(const v16h*)(kr + 32);   // e = 32 + 16*laneHi + 0..15
      v8f z{};
      v8f c = __builtin_amdgcn_wmma_f32_16x16x32_f16(false, aQ[0], false, bK0,
                                                     (short)0, z, false, false);
      c     = __builtin_amdgcn_wmma_f32_16x16x32_f16(false, aQ[1], false, bK1,
                                                     (short)0, c, false, false);
      sc[sub] = c;
    }

    // ---- mask (s<=l -> -1e9) only near the diagonal; scale AFTER mask ----
    float x[2][8];
    if (s0 < l0b + 128) {               // wave-uniform branch
      #pragma unroll
      for (int sub = 0; sub < 2; ++sub) {
        const int s_col = s0 + sub * 16 + n16;
        #pragma unroll
        for (int j = 0; j < 8; ++j) {
          const int l_row = l0 + j + 8 * laneHi;
          x[sub][j] = ((s_col <= l_row) ? NEGF : sc[sub][j]) * SCALE;
        }
      }
    } else {                            // strictly past the diagonal: no mask
      #pragma unroll
      for (int sub = 0; sub < 2; ++sub)
        #pragma unroll
        for (int j = 0; j < 8; ++j) x[sub][j] = sc[sub][j] * SCALE;
    }

    // ---- online softmax ----
    float p[2][8], alpha[8];
    #pragma unroll
    for (int j = 0; j < 8; ++j) {
      const float rm = rowmax16(fmaxf(x[0][j], x[1][j]));
      const float mn = fmaxf(m_i[j], rm);
      alpha[j] = __expf(m_i[j] - mn);
      m_i[j]   = mn;
      p[0][j]  = __expf(x[0][j] - mn);
      p[1][j]  = __expf(x[1][j] - mn);
      l_i[j]   = l_i[j] * alpha[j] + rowsum16(p[0][j] + p[1][j]);
    }
    #pragma unroll
    for (int c = 0; c < 4; ++c)
      #pragma unroll
      for (int j = 0; j < 8; ++j) O[c][j] *= alpha[j];

    // ---- P: C-layout -> A-layout transpose via per-wave LDS (f16) ----
    #pragma unroll
    for (int sub = 0; sub < 2; ++sub)
      #pragma unroll
      for (int j = 0; j < 8; ++j) {
        const int row = j + 8 * laneHi;
        myP[row * LDS_STRIDE + sub * 16 + n16] = (_Float16)p[sub][j];
      }
    asm volatile("s_wait_dscnt 0x0" ::: "memory");
    v16h aP;
    {
      const int off = 16 * laneHi;
      const _Float16* pr = myP + n16 * LDS_STRIDE;
      #pragma unroll
      for (int i = 0; i < 8; ++i) aP[i]     = pr[off + i];
      #pragma unroll
      for (int i = 0; i < 8; ++i) aP[8 + i] = pr[8 + off + i];
    }

    // ---- O += P(16x32) x V(32x16) from transposed V tile in LDS ----
    #pragma unroll
    for (int c = 0; c < 4; ++c) {
      const _Float16* vr = vT + (size_t)(16 * c + n16) * 32 + 16 * laneHi;
      v16h bV = *(const v16h*)vr;           // k = 16*laneHi + 0..15
      O[c] = __builtin_amdgcn_wmma_f32_16x16x32_f16(false, aP, false, bV,
                                                    (short)0, O[c], false, false);
    }
    cur ^= 1;
  }

  // ---- normalize and store fp32 ----
  #pragma unroll
  for (int j = 0; j < 8; ++j) {
    const float inv = 1.0f / l_i[j];
    const int   row = j + 8 * laneHi;
    float* orow = Ob + (size_t)(l0 + row) * H * D;
    #pragma unroll
    for (int c = 0; c < 4; ++c)
      orow[16 * c + n16] = O[c][j] * inv;
  }
}

// ---------------------------------------------------------------------------
// Fallback (no workspace): direct-fp32 kernel, one wave per 16 rows
// ---------------------------------------------------------------------------
__global__ __launch_bounds__(256) void flash_attn_direct(
    const float* __restrict__ Q, const float* __restrict__ K,
    const float* __restrict__ V, float* __restrict__ out)
{
  constexpr int B = 2, L = 2048, H = 16, E = 64, D = 64, S = 2048;
  constexpr float SCALE = 0.125f, NEGF = -1.0e9f;

  __shared__ _Float16 ldsP[8][16 * LDS_STRIDE];

  const int lane = threadIdx.x & 31;
  const int wave = threadIdx.x >> 5;
  const int tile = blockIdx.x * 8 + wave;
  const int ltiles = L / 16;
  if (tile >= B * H * ltiles) return;

  const int l0 = (tile % ltiles) * 16;
  const int h  = (tile / ltiles) % H;
  const int b  =  tile / (ltiles * H);
  const int laneHi = lane >> 4, n16 = lane & 15;
  const int rs = H * E;

  const float* Qb = Q + (size_t)b * L * H * E + (size_t)h * E;
  const float* Kb = K + (size_t)b * S * H * E + (size_t)h * E;
  const float* Vb = V + (size_t)b * S * H * D + (size_t)h * D;
  float*       Ob = out + (size_t)b * L * H * D + (size_t)h * D;

  v16h aQ[2];
  {
    const float* qr = Qb + (size_t)(l0 + n16) * rs;
    #pragma unroll
    for (int eh = 0; eh < 2; ++eh) {
      const int e0 = 32 * eh + 16 * laneHi;
      #pragma unroll
      for (int i = 0; i < 8; ++i) aQ[eh][i]     = (_Float16)qr[e0 + i];
      #pragma unroll
      for (int i = 0; i < 8; ++i) aQ[eh][8 + i] = (_Float16)qr[e0 + 8 + i];
    }
  }
  v8f O[4] = {v8f{}, v8f{}, v8f{}, v8f{}};
  float m_i[8], l_i[8];
  #pragma unroll
  for (int j = 0; j < 8; ++j) { m_i[j] = -3.0e38f; l_i[j] = 0.0f; }
  _Float16* myP = &ldsP[wave][0];

  const int s_start = (l0 == L - 16) ? 0 : (l0 & ~31);

  for (int s0 = s_start; s0 < S; s0 += 32) {
    v8f sc[2];
    #pragma unroll
    for (int sub = 0; sub < 2; ++sub) {
      const float* kr = Kb + (size_t)(s0 + sub * 16 + n16) * rs + 16 * laneHi;
      v16h bK0, bK1;
      #pragma unroll
      for (int i = 0; i < 16; ++i) bK0[i] = (_Float16)kr[i];
      #pragma unroll
      for (int i = 0; i < 16; ++i) bK1[i] = (_Float16)kr[32 + i];
      v8f z{};
      v8f c = __builtin_amdgcn_wmma_f32_16x16x32_f16(false, aQ[0], false, bK0,
                                                     (short)0, z, false, false);
      c     = __builtin_amdgcn_wmma_f32_16x16x32_f16(false, aQ[1], false, bK1,
                                                     (short)0, c, false, false);
      sc[sub] = c;
    }
    float x[2][8];
    if (s0 < l0 + 16) {
      #pragma unroll
      for (int sub = 0; sub < 2; ++sub) {
        const int s_col = s0 + sub * 16 + n16;
        #pragma unroll
        for (int j = 0; j < 8; ++j) {
          const int l_row = l0 + j + 8 * laneHi;
          x[sub][j] = ((s_col <= l_row) ? NEGF : sc[sub][j]) * SCALE;
        }
      }
    } else {
      #pragma unroll
      for (int sub = 0; sub < 2; ++sub)
        #pragma unroll
        for (int j = 0; j < 8; ++j) x[sub][j] = sc[sub][j] * SCALE;
    }
    float p[2][8], alpha[8];
    #pragma unroll
    for (int j = 0; j < 8; ++j) {
      const float rm = rowmax16(fmaxf(x[0][j], x[1][j]));
      const float mn = fmaxf(m_i[j], rm);
      alpha[j] = __expf(m_i[j] - mn);
      m_i[j]   = mn;
      p[0][j]  = __expf(x[0][j] - mn);
      p[1][j]  = __expf(x[1][j] - mn);
      l_i[j]   = l_i[j] * alpha[j] + rowsum16(p[0][j] + p[1][j]);
    }
    #pragma unroll
    for (int c = 0; c < 4; ++c)
      #pragma unroll
      for (int j = 0; j < 8; ++j) O[c][j] *= alpha[j];
    #pragma unroll
    for (int sub = 0; sub < 2; ++sub)
      #pragma unroll
      for (int j = 0; j < 8; ++j)
        myP[(j + 8 * laneHi) * LDS_STRIDE + sub * 16 + n16] = (_Float16)p[sub][j];
    asm volatile("s_wait_dscnt 0x0" ::: "memory");
    v16h aP;
    {
      const int off = 16 * laneHi;
      const _Float16* pr = myP + n16 * LDS_STRIDE;
      #pragma unroll
      for (int i = 0; i < 8; ++i) aP[i]     = pr[off + i];
      #pragma unroll
      for (int i = 0; i < 8; ++i) aP[8 + i] = pr[8 + off + i];
    }
    #pragma unroll
    for (int c = 0; c < 4; ++c) {
      const float* vr = Vb + (size_t)(s0 + 16 * laneHi) * rs + 16 * c + n16;
      v16h bV;
      #pragma unroll
      for (int i = 0; i < 16; ++i) bV[i] = (_Float16)vr[(size_t)i * rs];
      O[c] = __builtin_amdgcn_wmma_f32_16x16x32_f16(false, aP, false, bV,
                                                    (short)0, O[c], false, false);
    }
  }
  #pragma unroll
  for (int j = 0; j < 8; ++j) {
    const float inv = 1.0f / l_i[j];
    float* orow = Ob + (size_t)(l0 + j + 8 * laneHi) * H * D;
    #pragma unroll
    for (int c = 0; c < 4; ++c) orow[16 * c + n16] = O[c][j] * inv;
  }
}

extern "C" void kernel_launch(void* const* d_in, const int* in_sizes, int n_in,
                              void* d_out, int out_size, void* d_ws, size_t ws_size,
                              hipStream_t stream) {
  (void)in_sizes; (void)n_in; (void)out_size;
  const float* Q = (const float*)d_in[0];
  const float* K = (const float*)d_in[1];
  const float* V = (const float*)d_in[2];
  float* out = (float*)d_out;

  constexpr size_t ELEMS = (size_t)2 * 16 * 2048 * 64;   // 4 Mi per tensor
  constexpr size_t NEED  = 2 * ELEMS * sizeof(_Float16); // 16 MiB

  if (ws_size >= NEED) {
    _Float16* Kh = (_Float16*)d_ws;
    _Float16* Vt = Kh + ELEMS;
    stage_kv_f16<<<(int)(ELEMS / 256), 256, 0, stream>>>(K, V, Kh, Vt);
    flash_attn_staged<<<512, 256, 0, stream>>>(Q, Kh, Vt, out);
  } else {
    flash_attn_direct<<<512, 256, 0, stream>>>(Q, K, V, out);
  }
}
